// T_LSTM_Transducer_47871705481276
// MI455X (gfx1250) — compile-verified, matrix operand order checked by reading, not run on hardware
//
#include <hip/hip_runtime.h>
#include <hip/hip_bf16.h>
#include <stdint.h>

#define Bn  4
#define Tn  120
#define Un  50
#define Vn  4232
#define DEn 256
#define DDn 512
#define Hn  2048
#define VPAD 4256
#define NTILES (VPAD/16)          /* 266 */
#define NPAIRS (NTILES/2)         /* 133 */
#define MTOT (Bn*Tn*(Un+1))       /* 24480 */

typedef __attribute__((ext_vector_type(8)))  float  v8f;
typedef __attribute__((ext_vector_type(8)))  __bf16 v8bf;
typedef __attribute__((ext_vector_type(16))) __bf16 v16bf;

static __device__ __forceinline__ unsigned short f2bf(float f) {
  unsigned int u = __float_as_uint(f);
  u += 0x7FFFu + ((u >> 16) & 1u);          // round-to-nearest-even
  return (unsigned short)(u >> 16);
}

static __device__ __forceinline__ float fast_tanh(float x) {
  float t = __expf(fminf(fmaxf(2.0f * x, -30.0f), 30.0f));
  return (t - 1.0f) / (t + 1.0f);
}

static __device__ __forceinline__ v16bf ld16(const unsigned short* p0,
                                             const unsigned short* p1) {
  v16bf r;
  ((v8bf*)&r)[0] = *(const v8bf*)p0;        // K = khalf*8 .. +7
  ((v8bf*)&r)[1] = *(const v8bf*)p1;        // K = 16+khalf*8 .. +7
  return r;
}

static __device__ __forceinline__ float rmax16(float v) {
  v = fmaxf(v, __shfl_xor(v, 1, 16));
  v = fmaxf(v, __shfl_xor(v, 2, 16));
  v = fmaxf(v, __shfl_xor(v, 4, 16));
  v = fmaxf(v, __shfl_xor(v, 8, 16));
  return v;
}
static __device__ __forceinline__ float rsum16(float v) {
  v += __shfl_xor(v, 1, 16);
  v += __shfl_xor(v, 2, 16);
  v += __shfl_xor(v, 4, 16);
  v += __shfl_xor(v, 8, 16);
  return v;
}

// ---- W_out (f32 [H][V]) -> bf16 W_out^T ([VPAD][H]), zero-padded rows ----
__global__ void k_wout_bf16_T(const float* __restrict__ W,
                              unsigned short* __restrict__ WT) {
  __shared__ float tile[32][33];
  int tx = threadIdx.x, ty = threadIdx.y;
  int v0 = blockIdx.x * 32, h0 = blockIdx.y * 32;
#pragma unroll
  for (int i = 0; i < 4; ++i) {
    int h = h0 + ty + i * 8;
    int v = v0 + tx;
    tile[ty + i * 8][tx] = (v < Vn) ? W[(size_t)h * Vn + v] : 0.0f;
  }
  __syncthreads();
#pragma unroll
  for (int i = 0; i < 4; ++i) {
    int v = v0 + ty + i * 8;
    int h = h0 + tx;
    WT[(size_t)v * Hn + h] = f2bf(tile[tx][ty + i * 8]);
  }
}

// ---- bias padded with -1e30 so padded columns vanish in logsumexp ----
__global__ void k_bias_pad(const float* __restrict__ b_out,
                           float* __restrict__ bp) {
  int i = blockIdx.x * blockDim.x + threadIdx.x;
  if (i < VPAD) bp[i] = (i < Vn) ? b_out[i] : -1.0e30f;
}

// ---- small fp32 projections: Y[rows][H] = X[rows][K] @ W[K][H] ----
__global__ void k_proj(const float* __restrict__ X, const float* __restrict__ W,
                       float* __restrict__ Y, int K) {
  int row = blockIdx.y;
  int col = blockIdx.x * blockDim.x + threadIdx.x;
  const float* x = X + (size_t)row * K;
  float acc = 0.0f;
  for (int k = 0; k < K; ++k) acc = fmaf(x[k], W[(size_t)k * Hn + col], acc);
  Y[(size_t)row * Hn + col] = acc;
}

// ---- fused joint-GEMM + online logsumexp over V ----
// block: 128 threads (4 waves), 32 cells (rows); each wave owns pairs of
// N-tiles (2x2 register blocking: 4 independent WMMA chains, 1 ds + 1 vmem
// load per WMMA).
__global__ void __launch_bounds__(128)
k_joint(const float* __restrict__ encp, const float* __restrict__ decp,
        const float* __restrict__ bj, const unsigned short* __restrict__ WT,
        const float* __restrict__ bvp, const int* __restrict__ targets,
        float* __restrict__ lpb, float* __restrict__ lpl) {
  extern __shared__ char smem[];
  unsigned short* hbuf = (unsigned short*)smem;          // 32*2048 bf16 = 128KB
  float* pm   = (float*)(smem + 32 * Hn * 2);            // [4][32]
  float* ps   = pm + 128;                                // [4][32]
  float* lp0s = ps + 128;                                // [32]
  float* lpts = lp0s + 32;                               // [32]
  int*   tgts = (int*)(lpts + 32);                       // [32]
  int*   eoff = tgts + 32;                               // [32]
  int*   doff = eoff + 32;                               // [32]

  const int tid = threadIdx.x;
  const int mb0 = blockIdx.x * 32;

  if (tid < 32) {
    int cell = mb0 + tid;
    int b = cell / (Tn * (Un + 1));
    int rem = cell - b * (Tn * (Un + 1));
    int t = rem / (Un + 1);
    int u = rem - t * (Un + 1);
    eoff[tid] = (b * Tn + t) * Hn;
    doff[tid] = (b * (Un + 1) + u) * Hn;
    tgts[tid] = (u < Un) ? targets[b * Un + u] : -1;
  }
  __syncthreads();

  // fill h tile: tanh(enc_p + dec_p + b_joint) -> bf16 in LDS
  for (int idx = tid; idx < 32 * Hn; idx += 128) {
    int r = idx >> 11;          // Hn == 2048
    int k = idx & (Hn - 1);
    float v = encp[eoff[r] + k] + decp[doff[r] + k] + bj[k];
    hbuf[idx] = f2bf(fast_tanh(v));
  }
  __syncthreads();

  const int lane  = tid & 31;
  const int wave  = tid >> 5;
  const int nrow  = lane & 15;
  const int nhalf = lane >> 4;

  // A layout (16-bit 16x32): lane&15 = M row; lane>>4 selects K swizzle half.
  const unsigned short* A0base = hbuf + (size_t)nrow * Hn + nhalf * 8;
  const unsigned short* A1base = hbuf + (size_t)(16 + nrow) * Hn + nhalf * 8;

  float m0s[8], s0s[8], m1s[8], s1s[8];
#pragma unroll
  for (int j = 0; j < 8; ++j) {
    m0s[j] = -3.0e38f; s0s[j] = 0.0f;
    m1s[j] = -3.0e38f; s1s[j] = 0.0f;
  }

  for (int ntp = wave; ntp < NPAIRS; ntp += 4) {
    int n0 = ntp * 32;                               // tile A cols n0..n0+15
    const unsigned short* BAbase = WT + (size_t)(n0 + nrow) * Hn + nhalf * 8;
    const unsigned short* BBbase = BAbase + (size_t)16 * Hn;   // tile B cols +16
    v8f c00 = {0.f, 0.f, 0.f, 0.f, 0.f, 0.f, 0.f, 0.f};  // rows 0-15  x tileA
    v8f c01 = c00;                                        // rows 0-15  x tileB
    v8f c10 = c00;                                        // rows 16-31 x tileA
    v8f c11 = c00;                                        // rows 16-31 x tileB
    for (int k = 0; k < Hn; k += 32) {
      v16bf a0 = ld16(A0base + k, A0base + k + 16);
      v16bf a1 = ld16(A1base + k, A1base + k + 16);
      v16bf b0 = ld16(BAbase + k, BAbase + k + 16);
      v16bf b1 = ld16(BBbase + k, BBbase + k + 16);
      c00 = __builtin_amdgcn_wmma_f32_16x16x32_bf16(false, a0, false, b0,
                                                    (short)0, c00, false, false);
      c10 = __builtin_amdgcn_wmma_f32_16x16x32_bf16(false, a1, false, b0,
                                                    (short)0, c10, false, false);
      c01 = __builtin_amdgcn_wmma_f32_16x16x32_bf16(false, a0, false, b1,
                                                    (short)0, c01, false, false);
      c11 = __builtin_amdgcn_wmma_f32_16x16x32_bf16(false, a1, false, b1,
                                                    (short)0, c11, false, false);
    }
    int colA = n0 + nrow;
    int colB = colA + 16;
    float biasA = bvp[colA];
    float biasB = bvp[colB];
#pragma unroll
    for (int j = 0; j < 8; ++j) {
      int r0 = j + 8 * nhalf;       // C layout: VGPR j, lanes>=16 -> M=8+j
      int r1 = 16 + r0;
      float vA0 = c00[j] + biasA;   // row r0, tileA
      float vB0 = c01[j] + biasB;   // row r0, tileB
      float vA1 = c10[j] + biasA;   // row r1, tileA
      float vB1 = c11[j] + biasB;   // row r1, tileB

      if (colA == 0) { lp0s[r0] = vA0; lp0s[r1] = vA1; }
      if (colA == tgts[r0]) lpts[r0] = vA0;
      if (colB == tgts[r0]) lpts[r0] = vB0;
      if (colA == tgts[r1]) lpts[r1] = vA1;
      if (colB == tgts[r1]) lpts[r1] = vB1;

      // combined online logsumexp update over the 32 columns of the pair
      float t0 = fmaxf(rmax16(vA0), rmax16(vB0));
      float q0 = rsum16(__expf(vA0 - t0)) + rsum16(__expf(vB0 - t0));
      float nm0 = fmaxf(m0s[j], t0);
      s0s[j] = s0s[j] * __expf(m0s[j] - nm0) + q0 * __expf(t0 - nm0);
      m0s[j] = nm0;

      float t1 = fmaxf(rmax16(vA1), rmax16(vB1));
      float q1 = rsum16(__expf(vA1 - t1)) + rsum16(__expf(vB1 - t1));
      float nm1 = fmaxf(m1s[j], t1);
      s1s[j] = s1s[j] * __expf(m1s[j] - nm1) + q1 * __expf(t1 - nm1);
      m1s[j] = nm1;
    }
  }

  if (nrow == 0) {
#pragma unroll
    for (int j = 0; j < 8; ++j) {
      pm[wave * 32 + 8 * nhalf + j] = m0s[j];
      ps[wave * 32 + 8 * nhalf + j] = s0s[j];
      pm[wave * 32 + 16 + 8 * nhalf + j] = m1s[j];
      ps[wave * 32 + 16 + 8 * nhalf + j] = s1s[j];
    }
  }
  __syncthreads();

  if (tid < 32) {
    float M = -3.0e38f;
#pragma unroll
    for (int w = 0; w < 4; ++w) M = fmaxf(M, pm[w * 32 + tid]);
    float S = 0.0f;
#pragma unroll
    for (int w = 0; w < 4; ++w) S += ps[w * 32 + tid] * __expf(pm[w * 32 + tid] - M);
    float lse = M + __logf(S);
    int cell = mb0 + tid;
    int b = cell / (Tn * (Un + 1));
    int rem = cell - b * (Tn * (Un + 1));
    int t = rem / (Un + 1);
    int u = rem - t * (Un + 1);
    lpb[cell] = lp0s[tid] - lse;
    if (u < Un) lpl[(b * Tn + t) * Un + u] = lpts[tid] - lse;
  }
}

// ---- RNN-T alpha recursion + final loss ----
__global__ void k_rnnt_dp(const float* __restrict__ lpb,
                          const float* __restrict__ lpl,
                          const int* __restrict__ tlen,
                          const int* __restrict__ ulen,
                          float* __restrict__ out) {
  __shared__ float al[Bn][Un + 1];
  __shared__ float lls[Bn];
  int b = threadIdx.x;
  if (b < Bn) {
    const float* LB = lpb + (size_t)b * Tn * (Un + 1);
    const float* LL = lpl + (size_t)b * Tn * Un;
    int tl = tlen[b], ul = ulen[b];
    float* a = al[b];
    a[0] = 0.0f;
    for (int u = 1; u <= Un; ++u) a[u] = a[u - 1] + LL[u - 1];
    float ll = 0.0f;
    if (tl - 1 == 0) ll = a[ul] + LB[ul];
    for (int t = 1; t < Tn; ++t) {
      float left = a[0] + LB[(t - 1) * (Un + 1)];
      a[0] = left;
      for (int u = 1; u <= Un; ++u) {
        float dg = a[u] + LB[(t - 1) * (Un + 1) + u];
        float hz = left + LL[t * Un + (u - 1)];
        float M = fmaxf(dg, hz);
        float mn = fminf(dg, hz);
        left = M + log1pf(__expf(mn - M));
        a[u] = left;
      }
      if (t == tl - 1) ll = a[ul] + LB[t * (Un + 1) + ul];
    }
    lls[b] = ll;
  }
  __syncthreads();
  if (threadIdx.x == 0) {
    float s = 0.0f;
    for (int i = 0; i < Bn; ++i) s += lls[i];
    out[0] = -s / (float)Bn;
  }
}

extern "C" void kernel_launch(void* const* d_in, const int* in_sizes, int n_in,
                              void* d_out, int out_size, void* d_ws, size_t ws_size,
                              hipStream_t stream) {
  const float* enc   = (const float*)d_in[0];
  const float* dec   = (const float*)d_in[1];
  const float* W_e   = (const float*)d_in[2];
  const float* W_d   = (const float*)d_in[3];
  const float* b_j   = (const float*)d_in[4];
  const float* W_out = (const float*)d_in[5];
  const float* b_out = (const float*)d_in[6];
  const int*   tgt   = (const int*)d_in[7];
  const int*   tlen  = (const int*)d_in[8];
  const int*   ulen  = (const int*)d_in[9];
  float* out = (float*)d_out;
  (void)in_sizes; (void)n_in; (void)out_size; (void)ws_size;

  char* ws = (char*)d_ws;
  size_t off = 0;
  unsigned short* WT = (unsigned short*)(ws + off); off += (size_t)VPAD * Hn * 2;
  float* encp = (float*)(ws + off);                 off += (size_t)Bn * Tn * Hn * 4;
  float* decp = (float*)(ws + off);                 off += (size_t)Bn * (Un + 1) * Hn * 4;
  float* bvp  = (float*)(ws + off);                 off += (size_t)VPAD * 4;
  float* lpb  = (float*)(ws + off);                 off += (size_t)MTOT * 4;
  float* lpl  = (float*)(ws + off);                 off += (size_t)Bn * Tn * Un * 4;

  size_t shmem = (size_t)32 * Hn * 2 + (128 + 128 + 32 + 32) * 4 + 32 * 4 * 3;
  (void)hipFuncSetAttribute((const void*)k_joint,
                            hipFuncAttributeMaxDynamicSharedMemorySize,
                            (int)shmem);

  k_wout_bf16_T<<<dim3(VPAD / 32, Hn / 32), dim3(32, 8), 0, stream>>>(W_out, WT);
  k_bias_pad<<<(VPAD + 255) / 256, 256, 0, stream>>>(b_out, bvp);
  k_proj<<<dim3(Hn / 256, Bn * Tn), 256, 0, stream>>>(enc, W_e, encp, DEn);
  k_proj<<<dim3(Hn / 256, Bn * (Un + 1)), 256, 0, stream>>>(dec, W_d, decp, DDn);
  k_joint<<<MTOT / 32, 128, shmem, stream>>>(encp, decp, b_j, WT, bvp, tgt, lpb, lpl);
  k_rnnt_dp<<<1, 32, 0, stream>>>(lpb, lpl, tlen, ulen, out);
}
